// ScMoFormer_39256001085528
// MI455X (gfx1250) — matrix operand order repaired and checked by reading, not compiled, not run on hardware
//
#include <hip/hip_runtime.h>
#include <hip/hip_bf16.h>

// ---------------------------------------------------------------------------
// ScMoFormer forward pass for MI455X (gfx1250, wave32, WMMA).
// GEMMs: v_wmma_f32_16x16x32_f16, 64x64 WG tiles, 4 waves x (2x2) 16x16 tiles.
// Scatter/aggregate: f32 global atomics (bandwidth-bound part of the model).
// ---------------------------------------------------------------------------

typedef __attribute__((ext_vector_type(16))) _Float16 v16h;
typedef __attribute__((ext_vector_type(8)))  _Float16 v8h;
typedef __attribute__((ext_vector_type(4)))  _Float16 v4h;
typedef __attribute__((ext_vector_type(8)))  float    v8f;

#define NCELL 8192
#define NGENE 4096
#define NPRO  140
#define FDIM  512
#define DDIM  256
#define LDS_STR 40   // f16 elements per LDS tile row (32 + 8 pad -> 80 B)

static inline int cdiv(int a, int b) { return (a + b - 1) / b; }

#define SHUF16(lo, hi) __builtin_shufflevector((lo), (hi), 0, 1, 2, 3, 4, 5, 6, 7, \
                                               8, 9, 10, 11, 12, 13, 14, 15)

// ---------------------------------------------------------------------------
// MODE 0:  C[M,N] = A[M,K] @ B[K,N] + bias[N]
// MODE 1:  C[M,N] += A[M,K] @ B[K,N]
// Block = 128 threads (4 waves). WG tile 64x64, K-step 32 (K % 32 == 0).
// Each wave computes a 32x32 C region = 4 WMMAs per K-step.
// ---------------------------------------------------------------------------
template <int MODE>
__global__ void __launch_bounds__(128)
wmma_gemm_kernel(const float* __restrict__ A, const float* __restrict__ B,
                 const float* __restrict__ bias, float* __restrict__ C,
                 int M, int N, int K) {
    __shared__ _Float16 la[64 * LDS_STR];   // A tile: [m][k]
    __shared__ _Float16 lb[64 * LDS_STR];   // B tile transposed: [n][k]

    const int t    = threadIdx.x;
    const int wave = t >> 5;
    const int lane = t & 31;
    const int hl   = lane >> 4;          // lane half (0/1)
    const int lrow = lane & 15;
    const int wm   = (wave >> 1) * 32;   // wave row offset in WG tile
    const int wn   = (wave & 1) * 32;    // wave col offset in WG tile
    const int m0   = blockIdx.y * 64;
    const int n0   = blockIdx.x * 64;

    // B-loader coordinates: each thread owns one column strip of 16 k values.
    const int  bn     = t & 63;
    const int  bkh    = (t >> 6) * 16;
    const int  gn_ld  = n0 + bn;
    const bool nvalid = (gn_ld < N);
    const int  ns     = nvalid ? gn_ld : (N - 1);

    v8f acc00 = {}, acc01 = {}, acc10 = {}, acc11 = {};

    for (int k0 = 0; k0 < K; k0 += 32) {
        // ---- stage A tile: 64x32 f32 -> f16, branch-free float4 loads ----
        #pragma unroll
        for (int i = 0; i < 4; ++i) {
            int idx = t + i * 128;
            int r   = idx >> 3;
            int c4  = (idx & 7) * 4;
            int gm  = m0 + r;
            int gs  = (gm < M) ? gm : (M - 1);     // clamp: load stays valid
            float4 v = *(const float4*)(A + (size_t)gs * K + k0 + c4);
            bool ok = (gm < M);
            v4h h;
            h[0] = (_Float16)(ok ? v.x : 0.0f);
            h[1] = (_Float16)(ok ? v.y : 0.0f);
            h[2] = (_Float16)(ok ? v.z : 0.0f);
            h[3] = (_Float16)(ok ? v.w : 0.0f);
            *(v4h*)(&la[r * LDS_STR + c4]) = h;
        }
        // ---- stage B tile transposed: lb[n][k], coalesced column strips ----
        {
            v8h b0, b1;
            #pragma unroll
            for (int j = 0; j < 8; ++j) {
                float x0 = B[(size_t)(k0 + bkh + j) * N + ns];
                float x1 = B[(size_t)(k0 + bkh + 8 + j) * N + ns];
                b0[j] = (_Float16)(nvalid ? x0 : 0.0f);
                b1[j] = (_Float16)(nvalid ? x1 : 0.0f);
            }
            *(v8h*)(&lb[bn * LDS_STR + bkh])     = b0;
            *(v8h*)(&lb[bn * LDS_STR + bkh + 8]) = b1;
        }
        __syncthreads();

        // ---- fragments (ISA 7.12.2 16-bit layouts), all b128 LDS loads ----
        const _Float16* ar0 = &la[(wm + lrow)      * LDS_STR + hl * 8];
        const _Float16* ar1 = &la[(wm + 16 + lrow) * LDS_STR + hl * 8];
        v16h af0 = SHUF16(*(const v8h*)(ar0), *(const v8h*)(ar0 + 16));
        v16h af1 = SHUF16(*(const v8h*)(ar1), *(const v8h*)(ar1 + 16));

        const _Float16* br0 = &lb[(wn + lrow)      * LDS_STR + hl * 8];
        const _Float16* br1 = &lb[(wn + 16 + lrow) * LDS_STR + hl * 8];
        v16h bf0 = SHUF16(*(const v8h*)(br0), *(const v8h*)(br0 + 16));
        v16h bf1 = SHUF16(*(const v8h*)(br1), *(const v8h*)(br1 + 16));

        acc00 = __builtin_amdgcn_wmma_f32_16x16x32_f16(false, af0, false, bf0,
                                                       (short)0, acc00, false, false);
        acc01 = __builtin_amdgcn_wmma_f32_16x16x32_f16(false, af0, false, bf1,
                                                       (short)0, acc01, false, false);
        acc10 = __builtin_amdgcn_wmma_f32_16x16x32_f16(false, af1, false, bf0,
                                                       (short)0, acc10, false, false);
        acc11 = __builtin_amdgcn_wmma_f32_16x16x32_f16(false, af1, false, bf1,
                                                       (short)0, acc11, false, false);
        __syncthreads();
    }

    // ---- epilogue: C/D layout -> VGPR i: row = hl*8 + i, col = lane%16 ----
    const int col0 = n0 + wn + lrow;
    const int col1 = col0 + 16;
    const int row0 = m0 + wm + hl * 8;
    const int row1 = row0 + 16;

    if (m0 + 64 <= M && n0 + 64 <= N) {
        // Interior block: fully unconditional, batched memory ops.
        float* p0 = C + (size_t)row0 * N;
        float* p1 = C + (size_t)row1 * N;
        if (MODE == 1) {
            float c00[8], c01[8], c10[8], c11[8];
            #pragma unroll
            for (int i = 0; i < 8; ++i) {
                c00[i] = p0[(size_t)i * N + col0];
                c01[i] = p0[(size_t)i * N + col1];
                c10[i] = p1[(size_t)i * N + col0];
                c11[i] = p1[(size_t)i * N + col1];
            }
            #pragma unroll
            for (int i = 0; i < 8; ++i) {
                p0[(size_t)i * N + col0] = acc00[i] + c00[i];
                p0[(size_t)i * N + col1] = acc01[i] + c01[i];
                p1[(size_t)i * N + col0] = acc10[i] + c10[i];
                p1[(size_t)i * N + col1] = acc11[i] + c11[i];
            }
        } else {
            const float bv0 = bias[col0];
            const float bv1 = bias[col1];
            #pragma unroll
            for (int i = 0; i < 8; ++i) {
                p0[(size_t)i * N + col0] = acc00[i] + bv0;
                p0[(size_t)i * N + col1] = acc01[i] + bv1;
                p1[(size_t)i * N + col0] = acc10[i] + bv0;
                p1[(size_t)i * N + col1] = acc11[i] + bv1;
            }
        }
    } else {
        // Edge block: per-element guards (only small / tail GEMMs hit this).
        const bool cv0 = (col0 < N), cv1 = (col1 < N);
        const float bv0 = (MODE == 0 && cv0) ? bias[col0] : 0.0f;
        const float bv1 = (MODE == 0 && cv1) ? bias[col1] : 0.0f;
        #pragma unroll
        for (int i = 0; i < 8; ++i) {
            if (row0 + i < M) {
                size_t r = (size_t)(row0 + i) * N;
                if (cv0) { float v = acc00[i] + (MODE ? C[r + col0] : bv0); C[r + col0] = v; }
                if (cv1) { float v = acc01[i] + (MODE ? C[r + col1] : bv1); C[r + col1] = v; }
            }
            if (row1 + i < M) {
                size_t r = (size_t)(row1 + i) * N;
                if (cv0) { float v = acc10[i] + (MODE ? C[r + col0] : bv0); C[r + col0] = v; }
                if (cv1) { float v = acc11[i] + (MODE ? C[r + col1] : bv1); C[r + col1] = v; }
            }
        }
    }
}

// ---------------------------------------------------------------------------
// Edge scatter: neigh[dst] += h_src[src] * w   (64 threads per edge, 4 dims)
// ---------------------------------------------------------------------------
__global__ void scatter_msgs_kernel(const float* __restrict__ h_src,
                                    const int* __restrict__ src,
                                    const int* __restrict__ dst,
                                    const float* __restrict__ w,
                                    float* __restrict__ neigh, int E) {
    int gid = blockIdx.x * blockDim.x + threadIdx.x;
    int e = gid >> 6;
    if (e >= E) return;
    int c = (gid & 63) * 4;
    int s = src[e], d = dst[e];
    float wv = w[e];
    float4 hv = *(const float4*)(h_src + (size_t)s * DDIM + c);
    float* nd = neigh + (size_t)d * DDIM + c;
    atomicAdd(nd + 0, hv.x * wv);
    atomicAdd(nd + 1, hv.y * wv);
    atomicAdd(nd + 2, hv.z * wv);
    atomicAdd(nd + 3, hv.w * wv);
}

__global__ void scatter_deg_kernel(const int* __restrict__ dst,
                                   float* __restrict__ deg, int E) {
    int e = blockIdx.x * blockDim.x + threadIdx.x;
    if (e < E) atomicAdd(&deg[dst[e]], 1.0f);
}

__global__ void normalize_neigh_kernel(float* __restrict__ neigh,
                                       const float* __restrict__ deg, int total) {
    int i = blockIdx.x * blockDim.x + threadIdx.x;
    if (i >= total) return;
    neigh[i] /= fmaxf(deg[i >> 8], 1.0f);
}

// ---------------------------------------------------------------------------
// Fused selu -> LayerNorm(D=256), optional accumulate into out.
// One block (256 threads) per row.
// ---------------------------------------------------------------------------
__global__ void __launch_bounds__(256)
selu_ln_kernel(const float* __restrict__ in, const float* __restrict__ g,
               const float* __restrict__ b, float* __restrict__ out, int add) {
    __shared__ float red[DDIM];
    const int row = blockIdx.x, t = threadIdx.x;
    float x = in[(size_t)row * DDIM + t];
    const float kScale = 1.0507009873554805f;
    const float kAlpha = 1.6732632423543772f;
    x = kScale * (x > 0.0f ? x : kAlpha * (expf(x) - 1.0f));

    red[t] = x;
    __syncthreads();
    for (int s = 128; s > 0; s >>= 1) {
        if (t < s) red[t] += red[t + s];
        __syncthreads();
    }
    float mu = red[0] * (1.0f / DDIM);
    __syncthreads();

    float d = x - mu;
    red[t] = d * d;
    __syncthreads();
    for (int s = 128; s > 0; s >>= 1) {
        if (t < s) red[t] += red[t + s];
        __syncthreads();
    }
    float var = red[0] * (1.0f / DDIM);

    float y = d * rsqrtf(var + 1e-5f) * g[t] + b[t];
    size_t o = (size_t)row * DDIM + t;
    if (add) out[o] += y; else out[o] = y;
}

// ---------------------------------------------------------------------------
// Small utility kernels
// ---------------------------------------------------------------------------
__global__ void gather_emb_kernel(const float* __restrict__ emb,
                                  const int* __restrict__ idx,
                                  float* __restrict__ out, int total) {
    int i = blockIdx.x * blockDim.x + threadIdx.x;
    if (i >= total) return;
    out[i] = emb[(size_t)idx[i >> 8] * DDIM + (i & 255)];
}

__global__ void copy_concat_kernel(const float* __restrict__ src,
                                   float* __restrict__ hist, int col0, int total) {
    int i = blockIdx.x * blockDim.x + threadIdx.x;
    if (i >= total) return;
    int r = i >> 8, c = i & 255;
    hist[(size_t)r * (3 * DDIM) + col0 + c] = src[i];
}

// ---------------------------------------------------------------------------
// Host-side helpers
// ---------------------------------------------------------------------------
static void launch_gemm(const float* A, const float* B, const float* bias,
                        float* C, int M, int N, int K, int accumulate,
                        hipStream_t stream) {
    dim3 grid(cdiv(N, 64), cdiv(M, 64));
    if (accumulate)
        wmma_gemm_kernel<1><<<grid, 128, 0, stream>>>(A, B, bias, C, M, N, K);
    else
        wmma_gemm_kernel<0><<<grid, 128, 0, stream>>>(A, B, bias, C, M, N, K);
}

static void run_sage(const float* h_src, const float* h_dst,
                     const int* src, const int* dst, const float* w,
                     int E, int n_dst,
                     const float* Ws, const float* Wn, const float* bs,
                     float* neigh, float* deg, float* sbuf, hipStream_t stream) {
    hipMemsetAsync(neigh, 0, (size_t)n_dst * DDIM * sizeof(float), stream);
    hipMemsetAsync(deg, 0, (size_t)n_dst * sizeof(float), stream);
    scatter_msgs_kernel<<<cdiv(E * 64, 256), 256, 0, stream>>>(h_src, src, dst, w, neigh, E);
    scatter_deg_kernel<<<cdiv(E, 256), 256, 0, stream>>>(dst, deg, E);
    normalize_neigh_kernel<<<cdiv(n_dst * DDIM, 256), 256, 0, stream>>>(neigh, deg, n_dst * DDIM);
    launch_gemm(h_dst, Ws, bs, sbuf, n_dst, DDIM, DDIM, 0, stream);      // self
    launch_gemm(neigh, Wn, nullptr, sbuf, n_dst, DDIM, DDIM, 1, stream); // + neigh
}

extern "C" void kernel_launch(void* const* d_in, const int* in_sizes, int n_in,
                              void* d_out, int out_size, void* d_ws, size_t ws_size,
                              hipStream_t stream) {
    (void)in_sizes; (void)n_in; (void)out_size; (void)ws_size;

    const float* cell_feat     = (const float*)d_in[0];
    const float* gene_feat     = (const float*)d_in[1];
    const int*   pro_idx       = (const int*)d_in[2];
    const int*   obs_src       = (const int*)d_in[3];
    const int*   obs_dst       = (const int*)d_in[4];
    const float* obs_w         = (const float*)d_in[5];
    const int*   rev_obs_src   = (const int*)d_in[6];
    const int*   rev_obs_dst   = (const int*)d_in[7];
    const float* rev_obs_w     = (const float*)d_in[8];
    const int*   symbol_src    = (const int*)d_in[9];
    const int*   symbol_dst    = (const int*)d_in[10];
    const float* symbol_w      = (const float*)d_in[11];
    const int*   rsym_src      = (const int*)d_in[12];
    const int*   rsym_dst      = (const int*)d_in[13];
    const float* rsym_w        = (const float*)d_in[14];
    const int*   coexp_src     = (const int*)d_in[15];
    const int*   coexp_dst     = (const int*)d_in[16];
    const float* coexp_w       = (const float*)d_in[17];
    const int*   ppi_src       = (const int*)d_in[18];
    const int*   ppi_dst       = (const int*)d_in[19];
    const float* ppi_w         = (const float*)d_in[20];
    const float* W_emb_cell    = (const float*)d_in[21];
    const float* b_emb_cell    = (const float*)d_in[22];
    const float* W_emb_gene    = (const float*)d_in[23];
    const float* b_emb_gene    = (const float*)d_in[24];
    const float* emb_pro       = (const float*)d_in[25];
    const float* W_mlpcell     = (const float*)d_in[26];
    const float* b_mlpcell     = (const float*)d_in[27];
    const float* W_sage_self   = (const float*)d_in[28];
    const float* W_sage_neigh  = (const float*)d_in[29];
    const float* b_sage        = (const float*)d_in[30];
    const float* ln1_g         = (const float*)d_in[31];
    const float* ln1_b         = (const float*)d_in[32];
    const float* ln2_g         = (const float*)d_in[33];
    const float* ln2_b         = (const float*)d_in[34];
    const float* W_last        = (const float*)d_in[35];
    const float* b_last        = (const float*)d_in[36];

    const int E_OBS = 300000, E_COEXP = 200000, E_PPI = 10000, E_SYM = 20000;

    // ---- workspace carving (floats) ----
    float* ws = (float*)d_ws;
    size_t o = 0;
    float* hc_a  = ws + o; o += (size_t)NCELL * DDIM;
    float* hc_b  = ws + o; o += (size_t)NCELL * DDIM;
    float* hg_a  = ws + o; o += (size_t)NGENE * DDIM;
    float* hg_b  = ws + o; o += (size_t)NGENE * DDIM;
    float* hp_a  = ws + o; o += (size_t)NPRO  * DDIM;
    float* hp_b  = ws + o; o += (size_t)NPRO  * DDIM;
    float* bufC  = ws + o; o += (size_t)NCELL * DDIM;
    float* bufG  = ws + o; o += (size_t)NGENE * DDIM;
    float* bufP  = ws + o; o += (size_t)NPRO  * DDIM;
    float* sbuf  = ws + o; o += (size_t)NCELL * DDIM;   // sage / mlp scratch
    float* neigh = ws + o; o += (size_t)NCELL * DDIM;   // max n_dst
    float* deg   = ws + o; o += (size_t)NCELL;
    float* hist  = ws + o; o += (size_t)NCELL * 3 * DDIM;

    // ---- embeddings ----
    launch_gemm(cell_feat, W_emb_cell, b_emb_cell, hc_a, NCELL, DDIM, FDIM, 0, stream);
    launch_gemm(gene_feat, W_emb_gene, b_emb_gene, hg_a, NGENE, DDIM, FDIM, 0, stream);
    gather_emb_kernel<<<cdiv(NPRO * DDIM, 256), 256, 0, stream>>>(
        emb_pro, pro_idx, hp_a, NPRO * DDIM);
    copy_concat_kernel<<<cdiv(NCELL * DDIM, 256), 256, 0, stream>>>(
        hc_a, hist, 0, NCELL * DDIM);

    float *hc = hc_a, *hcn = hc_b;
    float *hg = hg_a, *hgn = hg_b;
    float *hp = hp_a, *hpn = hp_b;

    for (int l = 0; l < 2; ++l) {
        const float* Ws_base = W_sage_self  + (size_t)l * 6 * DDIM * DDIM;
        const float* Wn_base = W_sage_neigh + (size_t)l * 6 * DDIM * DDIM;
        const float* bs_base = b_sage       + (size_t)l * 6 * DDIM;
        #define WS(r) (Ws_base + (size_t)(r) * DDIM * DDIM)
        #define WN(r) (Wn_base + (size_t)(r) * DDIM * DDIM)
        #define BS(r) (bs_base + (size_t)(r) * DDIM)
        #define LN1G(k) (ln1_g + (size_t)(l * 3 + (k)) * DDIM)
        #define LN1B(k) (ln1_b + (size_t)(l * 3 + (k)) * DDIM)
        #define LN2G(k) (ln2_g + (size_t)(l * 3 + (k)) * DDIM)
        #define LN2B(k) (ln2_b + (size_t)(l * 3 + (k)) * DDIM)

        // hmlp = LN1c(selu(hc @ Wmlp + bmlp))  -> bufC (add=0)
        launch_gemm(hc, W_mlpcell + (size_t)l * DDIM * DDIM,
                    b_mlpcell + (size_t)l * DDIM, sbuf, NCELL, DDIM, DDIM, 0, stream);
        selu_ln_kernel<<<NCELL, 256, 0, stream>>>(sbuf, LN1G(2), LN1B(2), bufC, 0);

        // obs (genes -> cells): bufC += LN1c(selu(sage))
        run_sage(hg, hc, obs_src, obs_dst, obs_w, E_OBS, NCELL,
                 WS(0), WN(0), BS(0), neigh, deg, sbuf, stream);
        selu_ln_kernel<<<NCELL, 256, 0, stream>>>(sbuf, LN1G(2), LN1B(2), bufC, 1);

        // rev_obs (cells -> genes): bufG = LN1g(selu(sage))
        run_sage(hc, hg, rev_obs_src, rev_obs_dst, rev_obs_w, E_OBS, NGENE,
                 WS(1), WN(1), BS(1), neigh, deg, sbuf, stream);
        selu_ln_kernel<<<NGENE, 256, 0, stream>>>(sbuf, LN1G(0), LN1B(0), bufG, 0);

        // rev_symbol (pro -> genes): bufG += LN1g(selu(sage))
        run_sage(hp, hg, rsym_src, rsym_dst, rsym_w, E_SYM, NGENE,
                 WS(3), WN(3), BS(3), neigh, deg, sbuf, stream);
        selu_ln_kernel<<<NGENE, 256, 0, stream>>>(sbuf, LN1G(0), LN1B(0), bufG, 1);

        // coexp (gene -> gene): bufG += LN1g(selu(sage))
        run_sage(hg, hg, coexp_src, coexp_dst, coexp_w, E_COEXP, NGENE,
                 WS(4), WN(4), BS(4), neigh, deg, sbuf, stream);
        selu_ln_kernel<<<NGENE, 256, 0, stream>>>(sbuf, LN1G(0), LN1B(0), bufG, 1);

        // symbol (genes -> pro): bufP = LN1p(selu(sage))
        run_sage(hg, hp, symbol_src, symbol_dst, symbol_w, E_SYM, NPRO,
                 WS(2), WN(2), BS(2), neigh, deg, sbuf, stream);
        selu_ln_kernel<<<NPRO, 256, 0, stream>>>(sbuf, LN1G(1), LN1B(1), bufP, 0);

        // ppi (pro -> pro): bufP += LN1p(selu(sage))
        run_sage(hp, hp, ppi_src, ppi_dst, ppi_w, E_PPI, NPRO,
                 WS(5), WN(5), BS(5), neigh, deg, sbuf, stream);
        selu_ln_kernel<<<NPRO, 256, 0, stream>>>(sbuf, LN1G(1), LN1B(1), bufP, 1);

        // final per-type LN2(selu(.)) into pong buffers
        selu_ln_kernel<<<NGENE, 256, 0, stream>>>(bufG, LN2G(0), LN2B(0), hgn, 0);
        selu_ln_kernel<<<NPRO,  256, 0, stream>>>(bufP, LN2G(1), LN2B(1), hpn, 0);
        selu_ln_kernel<<<NCELL, 256, 0, stream>>>(bufC, LN2G(2), LN2B(2), hcn, 0);

        copy_concat_kernel<<<cdiv(NCELL * DDIM, 256), 256, 0, stream>>>(
            hcn, hist, (l + 1) * DDIM, NCELL * DDIM);

        float* tmp;
        tmp = hc; hc = hcn; hcn = tmp;
        tmp = hg; hg = hgn; hgn = tmp;
        tmp = hp; hp = hpn; hpn = tmp;

        #undef WS
        #undef WN
        #undef BS
        #undef LN1G
        #undef LN1B
        #undef LN2G
        #undef LN2B
    }

    // head: out = hist[8192,768] @ W_last[768,140] + b_last
    launch_gemm(hist, W_last, b_last, (float*)d_out, NCELL, NPRO, 3 * DDIM, 0, stream);
}
